// RingDilatedAttentionHilbertCore_65755949301791
// MI455X (gfx1250) — matrix operand order repaired and checked by reading, not compiled, not run on hardware
//
#include <hip/hip_runtime.h>
#include <cstdint>
#include <cstddef>

#define DIM 512
#define HEADS 8
#define HEAD_DIM 64
#define SEQ 14336
#define KVLEN 2048
#define QK_SCALE 0.125f  // 1/sqrt(64), exact power of two

typedef __bf16 bf16_t;
typedef __attribute__((ext_vector_type(16))) __bf16 v16bf;
typedef __attribute__((ext_vector_type(8)))  __bf16 v8bf;
typedef __attribute__((ext_vector_type(8)))  float  v8f;

// ---- helpers -------------------------------------------------------------

static __device__ inline bf16_t f2bf(float f) {
  union { float f; uint32_t u; } x; x.f = f;
  uint32_t r = x.u + 0x7fffu + ((x.u >> 16) & 1u);  // round-to-nearest-even
  union { uint16_t u; bf16_t b; } y; y.u = (uint16_t)(r >> 16);
  return y.b;
}

static __device__ inline v16bf cat16(v8bf lo, v8bf hi) {
  v16bf r;
#pragma unroll
  for (int i = 0; i < 8; ++i) { r[i] = lo[i]; r[i + 8] = hi[i]; }
  return r;
}

// CDNA5 async global->LDS copy (16B per lane), tracked by ASYNCcnt.
// VDST = VGPR holding LDS byte offset, VADDR = 64-bit global address, SADDR=off.
static __device__ inline void async_cp16(uint32_t lds_off, uint64_t gaddr) {
  asm volatile("global_load_async_to_lds_b128 %0, %1, off"
               :: "v"(lds_off), "v"(gaddr) : "memory");
}
static __device__ inline void wait_async0() {
  asm volatile("s_wait_asynccnt 0" ::: "memory");
}
static __device__ inline uint32_t lds_off32(const void* p) {
  return (uint32_t)(uintptr_t)p;  // flat LDS addr: low 32 bits = LDS byte offset
}

// ---- fp32 -> bf16 conversion ---------------------------------------------

__global__ __launch_bounds__(256)
void k_f32_to_bf16(const float* __restrict__ src, bf16_t* __restrict__ dst, int n) {
  int i = blockIdx.x * 256 + threadIdx.x;
  if (i < n) dst[i] = f2bf(src[i]);
}

// ---- shared 128x128 bf16 WMMA GEMM core ----------------------------------
// C tile 128(M) x 128(N).  8 waves arranged 4(M) x 2(N); each wave computes
// 32(M) x 64(N) = 2x4 tiles of 16x16 via V_WMMA_F32_16X16X32_BF16.
// lds_a: A tile row-major [128][32] bf16 (async-staged).
// lds_b: B tile transposed [128n][32k] bf16 (VGPR-bounced transpose).

static __device__ inline void gemm_tile(
    const bf16_t* __restrict__ A, int lda,
    const bf16_t* __restrict__ B, int ldb,
    int m0, int n0, int K,
    bf16_t* lds_a, bf16_t* lds_b,
    v8f acc[2][4])
{
  const int t    = threadIdx.x;
  const int lane = t & 31, wid = t >> 5;
  const int wm   = wid & 3, wn  = wid >> 2;
  const int l16  = lane & 15, lh = lane >> 4;

#pragma unroll
  for (int mt = 0; mt < 2; ++mt)
#pragma unroll
    for (int nt = 0; nt < 4; ++nt)
#pragma unroll
      for (int r = 0; r < 8; ++r) acc[mt][nt][r] = 0.0f;

  for (int k0 = 0; k0 < K; k0 += 32) {
    __syncthreads();
    // Stage A asynchronously: 128 rows x 32 k (64 B/row); 512 x 16B chunks.
#pragma unroll
    for (int i = 0; i < 2; ++i) {
      int c = t + i * 256;
      int row = c >> 2, off = (c & 3) << 3;
      async_cp16(lds_off32(lds_a + row * 32 + off),
                 (uint64_t)(uintptr_t)(A + (size_t)(m0 + row) * lda + k0 + off));
    }
    // Stage B transposed (VGPR bounce): element (k, n) -> lds_b[n*32 + k].
#pragma unroll
    for (int i = 0; i < 2; ++i) {
      int c = t + i * 256;                 // 512 chunks of 8 contiguous n
      int k = c >> 4, ns = (c & 15) << 3;
      v8bf bv = *(const v8bf*)(B + (size_t)(k0 + k) * ldb + n0 + ns);
#pragma unroll
      for (int j = 0; j < 8; ++j) lds_b[(ns + j) * 32 + k] = bv[j];
    }
    if (k0 + 32 < K)  // pull next A tile toward the caches (global_prefetch_b8)
      __builtin_prefetch(A + (size_t)(m0 + (t >> 1)) * lda + k0 + 32, 0, 0);
    wait_async0();
    __syncthreads();

    v16bf af[2], bfrag[4];
#pragma unroll
    for (int mt = 0; mt < 2; ++mt) {
      // A 16x32 layout: lanes 0-15 K={0..7,16..23}, lanes 16-31 K={8..15,24..31}
      const bf16_t* arow = lds_a + (wm * 32 + mt * 16 + l16) * 32;
      af[mt] = cat16(*(const v8bf*)(arow + lh * 8),
                     *(const v8bf*)(arow + 16 + lh * 8));
    }
#pragma unroll
    for (int nt = 0; nt < 4; ++nt) {
      // B 32x16 layout: lanes 0-15 K={0..7,16..23}, lanes 16-31 K={8..15,24..31}, N = lane%16
      const bf16_t* brow = lds_b + (wn * 64 + nt * 16 + l16) * 32;
      bfrag[nt] = cat16(*(const v8bf*)(brow + lh * 8), *(const v8bf*)(brow + 16 + lh * 8));
    }
#pragma unroll
    for (int mt = 0; mt < 2; ++mt)
#pragma unroll
      for (int nt = 0; nt < 4; ++nt)
        acc[mt][nt] = __builtin_amdgcn_wmma_f32_16x16x32_bf16(
            false, af[mt], false, bfrag[nt], (short)0, acc[mt][nt], false, false);
  }
}

// ---- QKV projection: (14336x512) @ (512x1536) + bias, scatter bf16 Q/K/V --

__global__ __launch_bounds__(256)
void k_qkv_gemm(const bf16_t* __restrict__ xb, const bf16_t* __restrict__ wqkvb,
                const float* __restrict__ b_qkv,
                bf16_t* __restrict__ qb, bf16_t* __restrict__ kb,
                bf16_t* __restrict__ vb)
{
  __shared__ bf16_t lds_a[128 * 32];
  __shared__ bf16_t lds_b[128 * 32];
  const int m0 = blockIdx.x * 128, n0 = blockIdx.y * 128;
  v8f acc[2][4];
  gemm_tile(xb, DIM, wqkvb, 3 * DIM, m0, n0, DIM, lds_a, lds_b, acc);

  const int lane = threadIdx.x & 31, wid = threadIdx.x >> 5;
  const int wm = wid & 3, wn = wid >> 2;
  const int l16 = lane & 15, lh = lane >> 4;
#pragma unroll
  for (int nt = 0; nt < 4; ++nt) {
    const int n = n0 + wn * 64 + nt * 16 + l16;
    const float bias = b_qkv[n];
    const int part = n >> 9;           // 0=q 1=k 2=v
    const int h = (n & 511) >> 6;
    const int d = n & 63;
    bf16_t* dst = (part == 0) ? qb : ((part == 1) ? kb : vb);
    const float scl = (part == 0) ? QK_SCALE : 1.0f;  // fold 1/sqrt(d) into Q
#pragma unroll
    for (int mt = 0; mt < 2; ++mt)
#pragma unroll
      for (int r = 0; r < 8; ++r) {
        const int m = m0 + wm * 32 + mt * 16 + lh * 8 + r;
        dst[((size_t)h * SEQ + m) * 64 + d] = f2bf((acc[mt][nt][r] + bias) * scl);
      }
  }
}

// ---- dilated flash attention ---------------------------------------------
// block = (head, 128 q rows); 8 waves, each owns 16 q rows.
// All segments reduce to 2048 KV rows (stride dil); 16 chunks of 128.
// K chunks double-buffered + async-staged; V transposed via VGPR bounce.

__global__ __launch_bounds__(256)
void k_attention(const bf16_t* __restrict__ qb, const bf16_t* __restrict__ kb,
                 const bf16_t* __restrict__ vb, bf16_t* __restrict__ attnb)
{
  __shared__ bf16_t k_lds[2][128 * 64];     // K chunk row-major [kv][d]   2x16 KB
  __shared__ bf16_t vt_lds[2][64 * 128];    // V chunk transposed [d][kv]  2x16 KB
  __shared__ bf16_t p_lds[8 * 16 * 128];    // per-wave P tiles [16][128]  32 KB

  const int h = blockIdx.x;
  const int q_abs = blockIdx.y * 128;
  int kbase, dil;
  if      (q_abs < 2048) { kbase = 0;    dil = 1; }
  else if (q_abs < 6144) { kbase = 2048; dil = 2; }
  else                   { kbase = 6144; dil = 4; }

  const int t = threadIdx.x;
  const int lane = t & 31, wid = t >> 5;
  const int l16 = lane & 15, lh = lane >> 4;
  const int q0w = q_abs + wid * 16;

  // Q A-fragments for the wave's 16 rows, kept in VGPRs for the whole kernel.
  const bf16_t* qrow = qb + ((size_t)h * SEQ + q0w + l16) * 64;
  v16bf aq[2];
#pragma unroll
  for (int b = 0; b < 2; ++b)
    aq[b] = cat16(*(const v8bf*)(qrow + b * 32 + lh * 8),
                  *(const v8bf*)(qrow + b * 32 + 16 + lh * 8));

  v8f acco[4];
  float m_run[8], l_run[8];
#pragma unroll
  for (int dt = 0; dt < 4; ++dt)
#pragma unroll
    for (int r = 0; r < 8; ++r) acco[dt][r] = 0.0f;
#pragma unroll
  for (int r = 0; r < 8; ++r) { m_run[r] = -3.0e38f; l_run[r] = 0.0f; }

  bf16_t* pw = p_lds + wid * (16 * 128);

  // K chunk: async copy, 2 threads per row, 4x16B each.
  auto stage_k = [&](int c, bf16_t* kl) {
    const int row = t >> 1, half = t & 1;
    const int skv = kbase + (c * 128 + row) * dil;
    const uint64_t g = (uint64_t)(uintptr_t)(kb + ((size_t)h * SEQ + skv) * 64 + half * 32);
    const uint32_t l = lds_off32(kl + row * 64 + half * 32);
#pragma unroll
    for (int j = 0; j < 4; ++j) async_cp16(l + j * 16, g + j * 16);
  };
  // V chunk transposed [d][kv]: VGPR bounce.
  auto stage_v = [&](int c, bf16_t* vl) {
#pragma unroll
    for (int i = 0; i < 4; ++i) {
      const int cc = t + i * 256;            // 1024 chunks of 8 d-elements
      const int kv = cc >> 3, ds2 = (cc & 7) << 3;
      const int skv = kbase + (c * 128 + kv) * dil;
      v8bf vv = *(const v8bf*)(vb + ((size_t)h * SEQ + skv) * 64 + ds2);
#pragma unroll
      for (int j = 0; j < 8; ++j) vl[(ds2 + j) * 128 + kv] = vv[j];
    }
  };

  // Prologue: stage chunk 0 into buffer 0.
  stage_k(0, k_lds[0]);
  stage_v(0, vt_lds[0]);
  wait_async0();

  for (int c = 0; c < 16; ++c) {
    const int buf = c & 1;
    __syncthreads();  // staged data for chunk c visible to all waves

    // Kick next K chunk into the other buffer while we compute.
    if (c + 1 < 16) stage_k(c + 1, k_lds[buf ^ 1]);

    const bf16_t* kl = k_lds[buf];
    const bf16_t* vl = vt_lds[buf];

    // S = Q * K^T  (Q pre-scaled by 1/sqrt(d)); 8 kv tiles of 16.
    v8f accs[8];
#pragma unroll
    for (int j = 0; j < 8; ++j) {
      v8f s;
#pragma unroll
      for (int r = 0; r < 8; ++r) s[r] = 0.0f;
#pragma unroll
      for (int b = 0; b < 2; ++b) {
        const bf16_t* krow = kl + (j * 16 + l16) * 64 + b * 32;
        v16bf bk = cat16(*(const v8bf*)(krow + lh * 8), *(const v8bf*)(krow + 16 + lh * 8));
        s = __builtin_amdgcn_wmma_f32_16x16x32_bf16(false, aq[b], false, bk,
                                                    (short)0, s, false, false);
      }
      accs[j] = s;
    }

    // Online softmax.  C layout: VGPR r -> row lh*8+r, lane -> column.
    float mc[8];
#pragma unroll
    for (int r = 0; r < 8; ++r) mc[r] = accs[0][r];
#pragma unroll
    for (int j = 1; j < 8; ++j)
#pragma unroll
      for (int r = 0; r < 8; ++r) mc[r] = fmaxf(mc[r], accs[j][r]);
#pragma unroll
    for (int msk = 1; msk <= 8; msk <<= 1)    // reduce within 16-lane half
#pragma unroll
      for (int r = 0; r < 8; ++r) mc[r] = fmaxf(mc[r], __shfl_xor(mc[r], msk, 32));

    float corr[8], sums[8];
#pragma unroll
    for (int r = 0; r < 8; ++r) {
      const float mn = fmaxf(m_run[r], mc[r]);
      corr[r] = __expf(m_run[r] - mn);
      m_run[r] = mn;
      sums[r] = 0.0f;
    }
#pragma unroll
    for (int j = 0; j < 8; ++j)
#pragma unroll
      for (int r = 0; r < 8; ++r) {
        const float p = __expf(accs[j][r] - m_run[r]);
        sums[r] += p;
        pw[(lh * 8 + r) * 128 + j * 16 + l16] = f2bf(p);
      }
#pragma unroll
    for (int msk = 1; msk <= 8; msk <<= 1)
#pragma unroll
      for (int r = 0; r < 8; ++r) sums[r] += __shfl_xor(sums[r], msk, 32);
#pragma unroll
    for (int r = 0; r < 8; ++r) l_run[r] = l_run[r] * corr[r] + sums[r];
#pragma unroll
    for (int dt = 0; dt < 4; ++dt)
#pragma unroll
      for (int r = 0; r < 8; ++r) acco[dt][r] *= corr[r];

    // Same-wave LDS RAW on the P tile: wait on DS counter before re-reading.
    asm volatile("s_wait_dscnt 0" ::: "memory");

    // O += P * V : 4 blocks of 32 kv x 4 tiles of 16 d.
#pragma unroll
    for (int kbk = 0; kbk < 4; ++kbk) {
      const bf16_t* prow = pw + l16 * 128 + kbk * 32;
      v16bf ap = cat16(*(const v8bf*)(prow + lh * 8),
                       *(const v8bf*)(prow + 16 + lh * 8));
#pragma unroll
      for (int dt = 0; dt < 4; ++dt) {
        const bf16_t* vrow = vl + (dt * 16 + l16) * 128 + kbk * 32;
        v16bf bv = cat16(*(const v8bf*)(vrow + lh * 8), *(const v8bf*)(vrow + 16 + lh * 8));
        acco[dt] = __builtin_amdgcn_wmma_f32_16x16x32_bf16(
            false, ap, false, bv, (short)0, acco[dt], false, false);
      }
    }

    // Stage next V chunk into the other buffer; then drain this wave's
    // outstanding async K loads before the top-of-loop barrier.
    if (c + 1 < 16) stage_v(c + 1, vt_lds[buf ^ 1]);
    wait_async0();
  }

  // Normalize and store bf16 into attn buffer laid out [s][h*64+d].
#pragma unroll
  for (int dt = 0; dt < 4; ++dt)
#pragma unroll
    for (int r = 0; r < 8; ++r) {
      const int s = q0w + lh * 8 + r;
      const int col = h * 64 + dt * 16 + l16;
      attnb[(size_t)s * DIM + col] = f2bf(acco[dt][r] / l_run[r]);
    }
}

// ---- output projection: (14336x512) @ (512x512) + bias -> fp32 -----------

__global__ __launch_bounds__(256)
void k_out_gemm(const bf16_t* __restrict__ attnb, const bf16_t* __restrict__ woutb,
                const float* __restrict__ b_out, float* __restrict__ out)
{
  __shared__ bf16_t lds_a[128 * 32];
  __shared__ bf16_t lds_b[128 * 32];
  const int m0 = blockIdx.x * 128, n0 = blockIdx.y * 128;
  v8f acc[2][4];
  gemm_tile(attnb, DIM, woutb, DIM, m0, n0, DIM, lds_a, lds_b, acc);

  const int lane = threadIdx.x & 31, wid = threadIdx.x >> 5;
  const int wm = wid & 3, wn = wid >> 2;
  const int l16 = lane & 15, lh = lane >> 4;
#pragma unroll
  for (int nt = 0; nt < 4; ++nt) {
    const int n = n0 + wn * 64 + nt * 16 + l16;
    const float bias = b_out[n];
#pragma unroll
    for (int mt = 0; mt < 2; ++mt)
#pragma unroll
      for (int r = 0; r < 8; ++r) {
        const int m = m0 + wm * 32 + mt * 16 + lh * 8 + r;
        out[(size_t)m * DIM + n] = acc[mt][nt][r] + bias;
      }
  }
}

// ---- launch ---------------------------------------------------------------

extern "C" void kernel_launch(void* const* d_in, const int* in_sizes, int n_in,
                              void* d_out, int out_size, void* d_ws, size_t ws_size,
                              hipStream_t stream) {
  const float* x     = (const float*)d_in[0];
  const float* w_qkv = (const float*)d_in[1];
  const float* b_qkv = (const float*)d_in[2];
  const float* w_out = (const float*)d_in[3];
  const float* b_out = (const float*)d_in[4];
  float* out = (float*)d_out;

  char* ws = (char*)d_ws;
  bf16_t* xb    = (bf16_t*)(ws + 0);          // 14,680,064 B
  bf16_t* wqkvb = (bf16_t*)(ws + 14680064);   //  1,572,864 B
  bf16_t* woutb = (bf16_t*)(ws + 16252928);   //    524,288 B
  bf16_t* qb    = (bf16_t*)(ws + 16777216);   // 14,680,064 B [h][s][64]
  bf16_t* kbuf  = (bf16_t*)(ws + 31457280);   // 14,680,064 B
  bf16_t* vbuf  = (bf16_t*)(ws + 46137344);   // 14,680,064 B
  bf16_t* attnb = (bf16_t*)(ws + 60817408);   // 14,680,064 B [s][h*64+d]

  k_f32_to_bf16<<<(SEQ * DIM + 255) / 256, 256, 0, stream>>>(x, xb, SEQ * DIM);
  k_f32_to_bf16<<<(DIM * 3 * DIM + 255) / 256, 256, 0, stream>>>(w_qkv, wqkvb, DIM * 3 * DIM);
  k_f32_to_bf16<<<(DIM * DIM + 255) / 256, 256, 0, stream>>>(w_out, woutb, DIM * DIM);

  k_qkv_gemm<<<dim3(SEQ / 128, (3 * DIM) / 128), 256, 0, stream>>>(
      xb, wqkvb, b_qkv, qb, kbuf, vbuf);
  k_attention<<<dim3(HEADS, SEQ / 128), 256, 0, stream>>>(qb, kbuf, vbuf, attnb);
  k_out_gemm<<<dim3(SEQ / 128, DIM / 128), 256, 0, stream>>>(attnb, woutb, b_out, out);
}
